// SAFull_76175539962632
// MI455X (gfx1250) — matrix operand order repaired and checked by reading, not compiled, not run on hardware
//
#include <hip/hip_runtime.h>
#include <hip/hip_bf16.h>

typedef __attribute__((ext_vector_type(16))) _Float16 v16h;
typedef __attribute__((ext_vector_type(2)))  __fp16   v2h;   // matches cvt_pkrtz
typedef __attribute__((ext_vector_type(8)))  float    v8f;

#define BATCH 16
#define NHEAD 8
#define DH    32
#define LEN   1024
#define CIN   256

// ---- VALU butterfly (v_permlane16_b32) inside each 16-lane row -------------
// selector nibble for lane i = i ^ mask  (lo = lanes 0-7, hi = lanes 8-15)
__device__ __forceinline__ float permxor(float x, int lo, int hi) {
    int xi = __float_as_int(x);
    int yi = __builtin_amdgcn_permlane16(xi, xi, lo, hi, false, false);
    return __int_as_float(yi);
}
__device__ __forceinline__ float rowmax16(float x) {
    x = fmaxf(x, permxor(x, (int)0x67452301, (int)0xEFCDAB89)); // ^1
    x = fmaxf(x, permxor(x, (int)0x54761032, (int)0xDCFE98BA)); // ^2
    x = fmaxf(x, permxor(x, (int)0x32107654, (int)0xBA98FEDC)); // ^4
    x = fmaxf(x, permxor(x, (int)0xFEDCBA98, (int)0x76543210)); // ^8
    return x;
}
__device__ __forceinline__ float rowsum16(float x) {
    x += permxor(x, (int)0x67452301, (int)0xEFCDAB89);
    x += permxor(x, (int)0x54761032, (int)0xDCFE98BA);
    x += permxor(x, (int)0x32107654, (int)0xBA98FEDC);
    x += permxor(x, (int)0xFEDCBA98, (int)0x76543210);
    return x;
}

// ---------------------------------------------------------------------------
// Phase 1: grouped 1x1 conv -> Q,K,V (f16) + sine positional encoding on K.
//   Qt, Kt : (b, h, l, d)                      -> contiguous d
//   Vt     : (b, h, d, l') key-interleaved l'  -> contiguous l'
// l' within each 32-key block: key j -> 2j (j<16), key 16+j -> 2j+1
// ---------------------------------------------------------------------------
__global__ void qkv_proj_kernel(const float* __restrict__ x,
                                const float* __restrict__ wq,
                                const float* __restrict__ wk,
                                const float* __restrict__ wv,
                                _Float16* __restrict__ Qt,
                                _Float16* __restrict__ Kt,
                                _Float16* __restrict__ Vt)
{
    __shared__ float sW[3][DH][DH];
    const int tid = threadIdx.x;
    const int h = blockIdx.y, b = blockIdx.z;
    const int l = blockIdx.x * blockDim.x + tid;

    for (int idx = tid; idx < 3 * DH * DH; idx += blockDim.x) {
        int m = idx / (DH * DH);
        int r = idx % (DH * DH);
        const float* w = (m == 0) ? wq : (m == 1) ? wk : wv;
        sW[m][r / DH][r % DH] = w[h * DH * DH + r];
    }
    __syncthreads();

    float xv[DH];
    #pragma unroll
    for (int ci = 0; ci < DH; ++ci)
        xv[ci] = x[((size_t)(b * CIN + h * DH + ci) * LEN) + l];

    const int bh = b * NHEAD + h;
    const int lp = (l & ~31) | (((l & 15) << 1) | ((l >> 4) & 1)); // interleave
    const float fl = (float)l;
    const float nl4 = -9.210340371976184f; // -ln(10000)

    #pragma unroll 4
    for (int o = 0; o < DH; ++o) {
        float q = 0.f, k = 0.f, v = 0.f;
        #pragma unroll
        for (int ci = 0; ci < DH; ++ci) {
            q = fmaf(sW[0][o][ci], xv[ci], q);
            k = fmaf(sW[1][o][ci], xv[ci], k);
            v = fmaf(sW[2][o][ci], xv[ci], v);
        }
        int c = h * DH + o;
        int j = c >> 1;
        float freq = __expf(nl4 * (float)(2 * j) * (1.0f / 256.0f));
        float ang = freq * fl;
        k += (c & 1) ? __cosf(ang) : __sinf(ang);

        Qt[((size_t)bh * LEN + l) * DH + o] = (_Float16)q;
        Kt[((size_t)bh * LEN + l) * DH + o] = (_Float16)k;
        Vt[((size_t)bh * DH + o) * LEN + lp] = (_Float16)v;
    }
}

// ---- online softmax over a 64-key chunk for one 16-query sub-tile ----------
__device__ __forceinline__ void online_softmax(
    const v8f& S0, const v8f& S1, const v8f& S2, const v8f& S3,
    float m8[8], float l8[8], v8f& O0, v8f& O1,
    v2h Pb[16][32], int col, int half)
{
    #pragma unroll
    for (int r = 0; r < 8; ++r) {
        float s0 = S0[r], s1 = S1[r], s2 = S2[r], s3 = S3[r];
        float mx = fmaxf(fmaxf(s0, s1), fmaxf(s2, s3));
        mx = rowmax16(mx);
        float mn = fmaxf(m8[r], mx);
        float al = __expf(m8[r] - mn);
        float p0 = __expf(s0 - mn);
        float p1 = __expf(s1 - mn);
        float p2 = __expf(s2 - mn);
        float p3 = __expf(s3 - mn);
        float rs = rowsum16((p0 + p1) + (p2 + p3));
        l8[r] = l8[r] * al + rs;
        m8[r] = mn;
        O0[r] *= al;
        O1[r] *= al;
        // pack (key j, key j+16) pairs -> interleaved layout, one b32 store
        Pb[r + 8 * half][col]      = __builtin_amdgcn_cvt_pkrtz(p0, p1);
        Pb[r + 8 * half][16 + col] = __builtin_amdgcn_cvt_pkrtz(p2, p3);
    }
}

// ---------------------------------------------------------------------------
// Phase 2: flash attention. One wave per (b,h,32-query tile), 64 keys / iter.
// Each loaded K fragment feeds 8 S-WMMAs; each V fragment feeds 4 PV-WMMAs
// (P for both 16-query sub-tiles staged in LDS before one shared V load),
// halving L2 K/V traffic vs 16-query tiles. Single-wave workgroups: barriers
// are hardware NOPs, LDS is wave-private.
// ---------------------------------------------------------------------------
__global__ void __launch_bounds__(32)
flash_attn_kernel(const _Float16* __restrict__ Qt,
                  const _Float16* __restrict__ Kt,
                  const _Float16* __restrict__ Vt,
                  float* __restrict__ out)
{
    __shared__ __align__(16) v2h Pbuf[2][16][32]; // 2 sub-tiles x 16 q x 64 keys

    const int lane = threadIdx.x & 31;
    const int col  = lane & 15;
    const int half = lane >> 4;

    const int bh = blockIdx.x >> 5;   // 32 query tiles (of 32) per (b,h)
    const int qt = blockIdx.x & 31;

    const size_t qk_base = (size_t)bh * LEN * DH;
    const size_t v_base  = (size_t)bh * DH * LEN;

    // two Q sub-tiles, A layout: lane=row q, d contiguous 16 per lane half
    v16h aq0, aq1;
    {
        const _Float16* qp0 = Qt + qk_base + (size_t)(qt * 32 + col) * DH;
        const _Float16* qp1 = qp0 + 16 * DH;
        #pragma unroll
        for (int e = 0; e < 8; ++e) {
            aq0[e]     = qp0[half * 16 + e];
            aq0[e + 8] = qp0[half * 16 + 8 + e];
            aq1[e]     = qp1[half * 16 + e];
            aq1[e + 8] = qp1[half * 16 + 8 + e];
        }
    }

    float m8a[8], l8a[8], m8b[8], l8b[8];
    v8f O0a = {}, O1a = {}, O0b = {}, O1b = {};
    #pragma unroll
    for (int r = 0; r < 8; ++r) {
        m8a[r] = -__builtin_inff(); l8a[r] = 0.f;
        m8b[r] = -__builtin_inff(); l8b[r] = 0.f;
    }

    for (int kb = 0; kb < LEN / 64; ++kb) {
        const int n0 = kb * 64;

        // ---- K fragments for four 16-key sub-blocks (shared by both tiles) --
        v16h bk0, bk1, bk2, bk3;
        const _Float16* kp = Kt + qk_base + (size_t)(n0 + col) * DH + half * 16;
        #pragma unroll
        for (int e = 0; e < 16; ++e) {
            bk0[e] = kp[e];
            bk1[e] = kp[16 * DH + e];
            bk2[e] = kp[32 * DH + e];
            bk3[e] = kp[48 * DH + e];
        }
        v8f z = {};
        v8f Sa0 = __builtin_amdgcn_wmma_f32_16x16x32_f16(false, aq0, false, bk0, (short)0, z, false, false);
        v8f Sa1 = __builtin_amdgcn_wmma_f32_16x16x32_f16(false, aq0, false, bk1, (short)0, z, false, false);
        v8f Sa2 = __builtin_amdgcn_wmma_f32_16x16x32_f16(false, aq0, false, bk2, (short)0, z, false, false);
        v8f Sa3 = __builtin_amdgcn_wmma_f32_16x16x32_f16(false, aq0, false, bk3, (short)0, z, false, false);
        v8f Sb0 = __builtin_amdgcn_wmma_f32_16x16x32_f16(false, aq1, false, bk0, (short)0, z, false, false);
        v8f Sb1 = __builtin_amdgcn_wmma_f32_16x16x32_f16(false, aq1, false, bk1, (short)0, z, false, false);
        v8f Sb2 = __builtin_amdgcn_wmma_f32_16x16x32_f16(false, aq1, false, bk2, (short)0, z, false, false);
        v8f Sb3 = __builtin_amdgcn_wmma_f32_16x16x32_f16(false, aq1, false, bk3, (short)0, z, false, false);

        online_softmax(Sa0, Sa1, Sa2, Sa3, m8a, l8a, O0a, O1a, Pbuf[0], col, half);
        online_softmax(Sb0, Sb1, Sb2, Sb3, m8b, l8b, O0b, O1b, Pbuf[1], col, half);
        __syncthreads(); // 1-wave workgroup: HW NOP, compiler ordering only

        // ---- two K=32 PV steps; V fragments shared by both query sub-tiles --
        #pragma unroll
        for (int t = 0; t < 2; ++t) {
            const _Float16* ppa = (const _Float16*)&Pbuf[0][col][16 * t];
            const _Float16* ppb = (const _Float16*)&Pbuf[1][col][16 * t];
            v16h apa, apb;
            #pragma unroll
            for (int e = 0; e < 8; ++e) {
                apa[e]     = ppa[half * 16 + e];
                apa[e + 8] = ppa[half * 16 + 8 + e];
                apb[e]     = ppb[half * 16 + e];
                apb[e + 8] = ppb[half * 16 + 8 + e];
            }
            v16h bv0, bv1;
            const _Float16* vp = Vt + v_base + (size_t)col * LEN + n0 + 32 * t + half * 16;
            #pragma unroll
            for (int e = 0; e < 16; ++e) {
                bv0[e] = vp[e];
                bv1[e] = vp[16 * LEN + e];
            }
            O0a = __builtin_amdgcn_wmma_f32_16x16x32_f16(false, apa, false, bv0, (short)0, O0a, false, false);
            O1a = __builtin_amdgcn_wmma_f32_16x16x32_f16(false, apa, false, bv1, (short)0, O1a, false, false);
            O0b = __builtin_amdgcn_wmma_f32_16x16x32_f16(false, apb, false, bv0, (short)0, O0b, false, false);
            O1b = __builtin_amdgcn_wmma_f32_16x16x32_f16(false, apb, false, bv1, (short)0, O1b, false, false);
        }
        __syncthreads();
    }

    // ---- epilogue: normalize, regroup to float4, store (B, C, H*W) fp32 ----
    const int b = bh >> 3, h = bh & 7;
    float v0[8], v1[8], v2[8], v3[8];
    #pragma unroll
    for (int r = 0; r < 8; ++r) {
        float ia = 1.0f / l8a[r];
        float ib = 1.0f / l8b[r];
        v0[r] = O0a[r] * ia;
        v1[r] = O1a[r] * ia;
        v2[r] = O0b[r] * ib;
        v3[r] = O1b[r] * ib;
    }
    const int qba = qt * 32 + 8 * half;       // sub-tile a rows
    const int qbb = qba + 16;                 // sub-tile b rows
    float* c0 = out + (size_t)(b * CIN + h * DH + col) * LEN;
    float* c1 = out + (size_t)(b * CIN + h * DH + 16 + col) * LEN;
    ((float4*)(c0 + qba))[0] = make_float4(v0[0], v0[1], v0[2], v0[3]);
    ((float4*)(c0 + qba))[1] = make_float4(v0[4], v0[5], v0[6], v0[7]);
    ((float4*)(c1 + qba))[0] = make_float4(v1[0], v1[1], v1[2], v1[3]);
    ((float4*)(c1 + qba))[1] = make_float4(v1[4], v1[5], v1[6], v1[7]);
    ((float4*)(c0 + qbb))[0] = make_float4(v2[0], v2[1], v2[2], v2[3]);
    ((float4*)(c0 + qbb))[1] = make_float4(v2[4], v2[5], v2[6], v2[7]);
    ((float4*)(c1 + qbb))[0] = make_float4(v3[0], v3[1], v3[2], v3[3]);
    ((float4*)(c1 + qbb))[1] = make_float4(v3[4], v3[5], v3[6], v3[7]);
}

// ---------------------------------------------------------------------------
extern "C" void kernel_launch(void* const* d_in, const int* in_sizes, int n_in,
                              void* d_out, int out_size, void* d_ws, size_t ws_size,
                              hipStream_t stream) {
    const float* x  = (const float*)d_in[0];
    const float* wq = (const float*)d_in[1];
    const float* wk = (const float*)d_in[2];
    const float* wv = (const float*)d_in[3];
    float* out = (float*)d_out;

    const size_t nQK = (size_t)BATCH * NHEAD * LEN * DH; // 4M f16 elems each
    _Float16* Qt = (_Float16*)d_ws;
    _Float16* Kt = Qt + nQK;
    _Float16* Vt = Kt + nQK;   // total 24 MB of workspace

    qkv_proj_kernel<<<dim3(LEN / 256, NHEAD, BATCH), 256, 0, stream>>>(
        x, wq, wk, wv, Qt, Kt, Vt);

    // one single-wave workgroup per 32-query tile
    flash_attn_kernel<<<dim3(BATCH * NHEAD * (LEN / 32)), 32, 0, stream>>>(
        Qt, Kt, Vt, out);
}
